// ImageVpnGenerator_32538672234549
// MI455X (gfx1250) — matrix operand order repaired and checked by reading, not compiled, not run on hardware
//
#include <hip/hip_runtime.h>
#include <hip/hip_bf16.h>

// ---------------------------------------------------------------------------
// MI455X (gfx1250, wave32) implementation.
// Pipeline: [cvt weights->bf16] -> [WMMA GEMM Q/K/V (bf16 out), double-buffered
//           async global->LDS tile staging] ->
//           [fused attention: Q tile via TDM tensor_load_to_lds,
//            QK^T via global_load_b128, softmax, PV] ->
//           [WMMA GEMM mu / |var| -> f32 d_out]
// Workspace: 5 bf16 weight mats + Qbf/Kbf/Vbf/attn_feat bf16 ≈ 237 MiB.
// ---------------------------------------------------------------------------

typedef __attribute__((ext_vector_type(16))) __bf16 v16bf;
typedef __attribute__((ext_vector_type(8)))  float  v8f;
typedef __attribute__((ext_vector_type(8)))  int    i32x8;
typedef __attribute__((ext_vector_type(4)))  int    i32x4;
typedef __attribute__((ext_vector_type(4)))  unsigned int u32x4;

#define DDIM 768
#define HDIM 768
#define BATCH 64
#define NQ  197
#define NQP 208   // NQ padded to 16
#define NKD 1024

__device__ __forceinline__ unsigned short f2bf(float f) {
  unsigned int u = __builtin_bit_cast(unsigned int, f);
  u += 0x7FFFu + ((u >> 16) & 1u);          // round-to-nearest-even
  return (unsigned short)(u >> 16);
}

__device__ __forceinline__ v8f wmma_bf16f32(v16bf a, v16bf b, v8f c) {
  // (neg_a, A, neg_b, B, c_mod, C, reuse_a, reuse_b)
  return __builtin_amdgcn_wmma_f32_16x16x32_bf16(false, a, false, b, (short)0, c,
                                                 false, false);
}

// LDS aperture addresses keep the LDS byte offset in the low 32 bits.
__device__ __forceinline__ unsigned lds_off(const void* p) {
  return (unsigned)(size_t)p;
}

// Async global->LDS copy of 16 bytes (ASYNCcnt-tracked, no VGPR data path).
__device__ __forceinline__ void async_b128(unsigned dst_lds, const void* src) {
  const unsigned long long ga = (unsigned long long)(size_t)src;
  asm volatile("global_load_async_to_lds_b128 %0, %1, off"
               :: "v"(dst_lds), "v"(ga) : "memory");
}
__device__ __forceinline__ void wait_async0() {
  asm volatile("s_wait_asynccnt 0x0" ::: "memory");
}

// A fragment: 16x32 bf16 tile starting at K-offset koff (koff multiple of 32).
// Lane L: M = L&15 ; halves follow ISA layout (VGPR v, g=L>>4):
//   K = (v<4?0:16) + (v%4)*2 + 8*g + {0,1}
__device__ __forceinline__ v16bf load_a_frag(const unsigned short* base,
                                             int rs_halves, int koff, int lane) {
  const unsigned int* p = (const unsigned int*)base;
  const int m = lane & 15, g = lane >> 4;
  const int rs2 = rs_halves >> 1;
  i32x8 d;
#pragma unroll
  for (int dw = 0; dw < 8; ++dw) {
    const int k = ((dw & 4) << 2) + ((dw & 3) << 1) + (g << 3) + koff;
    d[dw] = p[m * rs2 + (k >> 1)];
  }
  return __builtin_bit_cast(v16bf, d);
}

// B fragment: 32x16 bf16. Lane L holds column N=n0+(L&15), K in A's table order.
__device__ __forceinline__ v16bf load_b_frag_lds(const unsigned short* Bs,
                                                 int rs_halves, int n0, int lane) {
  const unsigned short* p = Bs + n0 + (lane & 15);
  const int g = lane >> 4;
  i32x8 d;
#pragma unroll
  for (int dw = 0; dw < 8; ++dw) {
    const int k = ((dw & 4) << 2) + ((dw & 3) << 1) + (g << 3);
    d[dw] = (int)((unsigned)p[k * rs_halves] | ((unsigned)p[(k + 1) * rs_halves] << 16));
  }
  return __builtin_bit_cast(v16bf, d);
}

__device__ __forceinline__ v16bf load_b_frag_global(const unsigned short* rowptr) {
  i32x8 d;
#pragma unroll
  for (int dw = 0; dw < 8; ++dw) {
    const int k = ((dw & 4) << 2) + ((dw & 3) << 1);
    d[dw] = (int)((unsigned)rowptr[k * HDIM] | ((unsigned)rowptr[(k + 1) * HDIM] << 16));
  }
  return __builtin_bit_cast(v16bf, d);
}

// ---------------------------------------------------------------------------
__global__ __launch_bounds__(256) void cvt_f32_bf16_kernel(
    const float* __restrict__ src, unsigned short* __restrict__ dst, int n) {
  const int i = blockIdx.x * 256 + threadIdx.x;
  if (i < n) dst[i] = f2bf(src[i]);
}

// ---------------------------------------------------------------------------
// C[b] = A[b] (MxK) * B (KxN) + bias ; 64x64 block tile, BK=32, 8 waves.
// Wave (wm,wn): 16-row strip x 32 cols -> two 16x16 WMMA accumulators.
// Double-buffered LDS: tile k+1 is async-DMA'd while WMMAs consume tile k.
// One barrier per k-step. Mld = allocated rows per batch (async addr clamp).
template <bool A_F32, bool OUT_BF16, bool DO_ABS>
__global__ __launch_bounds__(256) void gemm_bias_kernel(
    const void* __restrict__ Abase, long long aBatchStride, int Mld,
    const unsigned short* __restrict__ B, const float* __restrict__ bias,
    void* __restrict__ Obase, long long oBatchStride, int M, int N, int K) {
  __shared__ alignas(16) unsigned short As[2][64 * 40];  // 64x32, stride 40
  __shared__ alignas(16) unsigned short Bs[2][32 * 72];  // 32x64, stride 72
  const int tid = threadIdx.x, lane = tid & 31, wave = tid >> 5;
  const int wm = wave & 3, wn = wave >> 2;
  const int bM = blockIdx.y * 64, bN = blockIdx.x * 64, bb = blockIdx.z;

  const int arow = tid >> 2, acg = (tid & 3) * 8;      // A tile: 8 halves/thread
  const int brow = tid >> 3, bcg = (tid & 7) * 8;      // B tile: 8 halves/thread
  const unsigned asDst[2] = {lds_off(&As[0][arow * 40 + acg]),
                             lds_off(&As[1][arow * 40 + acg])};
  const unsigned bsDst[2] = {lds_off(&Bs[0][brow * 72 + bcg]),
                             lds_off(&Bs[1][brow * 72 + bcg])};
  const int grow = bM + arow;

  auto stage = [&](int k0, int buf) {
    // B tile (32x64 bf16 = 4KB), one async b128 per lane
    async_b128(bsDst[buf], B + (long long)(k0 + brow) * N + bN + bcg);
    if (A_F32) {
      const float* A = (const float*)Abase + (long long)bb * aBatchStride;
      float4 x0 = {0.f, 0.f, 0.f, 0.f}, x1 = {0.f, 0.f, 0.f, 0.f};
      if (grow < M) {
        const float4* pa = (const float4*)(A + (long long)grow * K + k0 + acg);
        x0 = pa[0]; x1 = pa[1];
      }
      unsigned short* s = &As[buf][arow * 40 + acg];
      s[0] = f2bf(x0.x); s[1] = f2bf(x0.y); s[2] = f2bf(x0.z); s[3] = f2bf(x0.w);
      s[4] = f2bf(x1.x); s[5] = f2bf(x1.y); s[6] = f2bf(x1.z); s[7] = f2bf(x1.w);
    } else {
      const unsigned short* A =
          (const unsigned short*)Abase + (long long)bb * aBatchStride;
      const int crow = grow < Mld ? grow : (Mld - 1);  // rows >= M unused anyway
      async_b128(asDst[buf], A + (long long)crow * K + k0 + acg);
    }
  };

  v8f acc0 = {0, 0, 0, 0, 0, 0, 0, 0};
  v8f acc1 = {0, 0, 0, 0, 0, 0, 0, 0};

  stage(0, 0);
  const int nsteps = K / 32;
  for (int i = 0; i < nsteps; ++i) {
    const int buf = i & 1;
    wait_async0();       // my stage(i) landed in LDS
    __syncthreads();     // everyone's stage(i) landed; buf^1 free to overwrite
    if (i + 1 < nsteps) stage((i + 1) * 32, buf ^ 1);  // overlaps WMMAs below

    const v16bf af = load_a_frag(&As[buf][wm * 16 * 40], 40, 0, lane);
    const v16bf b0 = load_b_frag_lds(Bs[buf], 72, wn * 32, lane);
    const v16bf b1 = load_b_frag_lds(Bs[buf], 72, wn * 32 + 16, lane);
    acc0 = wmma_bf16f32(af, b0, acc0);
    acc1 = wmma_bf16f32(af, b1, acc1);
  }

  // epilogue: C layout lane L -> N = L&15, row = r + 8*(L>>4)
  const int g = lane >> 4, nlo = lane & 15;
  const int col0 = bN + wn * 32 + nlo;
  const float bv0 = bias[col0], bv1 = bias[col0 + 16];
#pragma unroll
  for (int r = 0; r < 8; ++r) {
    const int row = bM + wm * 16 + r + 8 * g;
    if (row < M) {
      float v0 = acc0[r] + bv0, v1 = acc1[r] + bv1;
      if (DO_ABS) { v0 = fabsf(v0); v1 = fabsf(v1); }
      if (OUT_BF16) {
        unsigned short* O =
            (unsigned short*)Obase + (long long)bb * oBatchStride;
        O[(long long)row * N + col0]      = f2bf(v0);
        O[(long long)row * N + col0 + 16] = f2bf(v1);
      } else {
        float* O = (float*)Obase + (long long)bb * oBatchStride;
        O[(long long)row * N + col0]      = v0;
        O[(long long)row * N + col0 + 16] = v1;
      }
    }
  }
}

// ---------------------------------------------------------------------------
// Fused attention per (batch, 16-row q tile): S = Q K^T ; softmax ; O = P V.
// Q tile is DMA'd into LDS by the Tensor Data Mover (one TENSOR_LOAD_TO_LDS).
// LDS: Qs 16x768 bf16 (24KB) | S 16x1032 f32 (66KB) | Ps 16x1032 bf16 (33KB)
//      | red 16x16 f32 (1KB)  -> ~123KB dynamic (WGP has 320KB).
__global__ __launch_bounds__(256) void attn_kernel(
    const unsigned short* __restrict__ Qbf, const unsigned short* __restrict__ Kbf,
    const unsigned short* __restrict__ Vbf, unsigned short* __restrict__ Abf) {
  extern __shared__ char smem[];
  unsigned short* Qs = (unsigned short*)smem;                          // 16*768
  float* S = (float*)(smem + 16 * 768 * 2);                            // 16*1032
  unsigned short* Ps =
      (unsigned short*)(smem + 16 * 768 * 2 + 16 * 1032 * 4);          // 16*1032
  float* red =
      (float*)(smem + 16 * 768 * 2 + 16 * 1032 * 4 + 16 * 1032 * 2);   // 16*16

  const int tid = threadIdx.x, lane = tid & 31, wave = tid >> 5;
  const int b = blockIdx.y, q0 = blockIdx.x * 16;

  // ---- TDM: DMA 16x768 bf16 Q tile into LDS (issued once, by wave 0).
  if (wave == 0) {
    const unsigned long long ga =
        (unsigned long long)(size_t)(Qbf + ((long long)b * NQP + q0) * HDIM);
    u32x4 g0;
    g0[0] = 1u;                                            // count=1 (valid D#)
    g0[1] = lds_off(Qs);                                   // lds_addr
    g0[2] = (unsigned)(ga & 0xFFFFFFFFu);                  // global_addr[31:0]
    g0[3] = (unsigned)((ga >> 32) & 0x01FFFFFFu)           // global_addr[56:32]
            | (2u << 30);                                  // type=2 ("image")
    i32x8 g1;
    g1[0] = (int)(1u << 16);                 // workgroup_mask=0, data_size=2B
    g1[1] = (int)((768u & 0xFFFFu) << 16);   // tensor_dim0[15:0]=768
    g1[2] = (int)((768u >> 16) | (16u << 16));   // dim0 hi | tensor_dim1=16
    g1[3] = (int)((16u >> 16) | (768u << 16));   // dim1 hi | tile_dim0=768
    g1[4] = 16;                              // tile_dim1=16, tile_dim2=0
    g1[5] = 768;                             // tensor_dim0_stride=768
    g1[6] = 0;
    g1[7] = 0;
    const i32x4 z4 = {0, 0, 0, 0};
    const i32x8 z8 = {0, 0, 0, 0, 0, 0, 0, 0};
    // clang-23 / therock-10.0 form: (g0, g1, g2, g3, g4, cpol)
    __builtin_amdgcn_tensor_load_to_lds(g0, g1, z4, z4, z8, 0);
  }
  __builtin_amdgcn_s_wait_tensorcnt(0);
  __syncthreads();

  // ---- Phase 1: scores. Each wave computes 8 of the 64 16x16 n-tiles.
  for (int i = 0; i < 8; ++i) {
    const int n0 = (wave * 8 + i) * 16;
    v8f acc = {0, 0, 0, 0, 0, 0, 0, 0};
    // per-lane base for transposed K fetch: lanes 0-15 -> rows n0+lane (cols
    // h0..h0+7), lanes 16-31 -> same rows, cols h0+8..h0+15 (b128 = 8 halves)
    const unsigned short* krow =
        Kbf + ((long long)b * NKD + n0 + (lane & 15)) * HDIM + 8 * (lane >> 4);
    const unsigned long long kaddr = (unsigned long long)(size_t)krow;
    for (int h0 = 0; h0 < HDIM; h0 += 32) {
      const v16bf aq = load_a_frag(Qs, HDIM, h0, lane);
      i32x4 lo, hi;
      const unsigned long long a0 = kaddr + (unsigned long long)h0 * 2ull;
      const unsigned long long a1 = a0 + 32ull;  // next 16 h-columns
      asm volatile("global_load_b128 %0, %1, off"
                   : "=v"(lo) : "v"(a0) : "memory");
      asm volatile("global_load_b128 %0, %1, off"
                   : "=v"(hi) : "v"(a1) : "memory");
      asm volatile("s_wait_loadcnt 0x0" ::: "memory");
      i32x8 bd;
      bd[0] = lo[0]; bd[1] = lo[1]; bd[2] = lo[2]; bd[3] = lo[3];
      bd[4] = hi[0]; bd[5] = hi[1]; bd[6] = hi[2]; bd[7] = hi[3];
      const v16bf bk = __builtin_bit_cast(v16bf, bd);
      acc = wmma_bf16f32(aq, bk, acc);
    }
    const int g = lane >> 4, nlo = lane & 15;
#pragma unroll
    for (int r = 0; r < 8; ++r) S[(r + 8 * g) * 1032 + n0 + nlo] = acc[r];
  }
  __syncthreads();

  // ---- Phase 2: softmax over NK (row r handled by 16 threads x 64 cols).
  {
    const int r = tid >> 4, seg = tid & 15;
    float* Srow = S + r * 1032 + seg * 64;
    float mx = -3.4e38f;
    for (int j = 0; j < 64; ++j) mx = fmaxf(mx, Srow[j]);
    red[r * 16 + seg] = mx;
    __syncthreads();
    float m = red[r * 16];
#pragma unroll
    for (int j = 1; j < 16; ++j) m = fmaxf(m, red[r * 16 + j]);
    __syncthreads();
    float sum = 0.f;
    for (int j = 0; j < 64; ++j) {
      const float e = __expf(Srow[j] - m);
      Srow[j] = e;
      sum += e;
    }
    red[r * 16 + seg] = sum;
    __syncthreads();
    float tot = 0.f;
#pragma unroll
    for (int j = 0; j < 16; ++j) tot += red[r * 16 + j];
    const float inv = 1.0f / tot;
    unsigned short* Prow = Ps + r * 1032 + seg * 64;
    for (int j = 0; j < 64; ++j) Prow[j] = f2bf(Srow[j] * inv);
  }
  __syncthreads();

  // ---- Phase 3: O = P V. Each wave owns 6 of the 48 16-col h tiles.
  for (int j = 0; j < 6; ++j) {
    const int h0 = (wave * 6 + j) * 16;
    v8f acc = {0, 0, 0, 0, 0, 0, 0, 0};
    for (int nk0 = 0; nk0 < NKD; nk0 += 32) {
      const v16bf ap = load_a_frag(Ps, 1032, nk0, lane);
      const unsigned short* vrow =
          Vbf + ((long long)b * NKD + nk0 + 8 * (lane >> 4)) * HDIM + h0 + (lane & 15);  // lane&15 = N col
      const v16bf bv = load_b_frag_global(vrow);
      acc = wmma_bf16f32(ap, bv, acc);
    }
    const int g = lane >> 4, nlo = lane & 15;
    unsigned short* Orow = Abf + ((long long)b * NQP + q0) * HDIM + h0 + nlo;
#pragma unroll
    for (int r = 0; r < 8; ++r)
      Orow[(long long)(r + 8 * g) * HDIM] = f2bf(acc[r]);
  }
}

// ---------------------------------------------------------------------------
extern "C" void kernel_launch(void* const* d_in, const int* in_sizes, int n_in,
                              void* d_out, int out_size, void* d_ws,
                              size_t ws_size, hipStream_t stream) {
  (void)in_sizes; (void)n_in; (void)out_size; (void)ws_size;
  const float* spatial = (const float*)d_in[0];
  const float* text    = (const float*)d_in[1];
  const float* Wq  = (const float*)d_in[2];  const float* bq   = (const float*)d_in[3];
  const float* Wk  = (const float*)d_in[4];  const float* bk   = (const float*)d_in[5];
  const float* Wv  = (const float*)d_in[6];  const float* bv   = (const float*)d_in[7];
  const float* Wvar = (const float*)d_in[8]; const float* bvar = (const float*)d_in[9];
  const float* Wmu  = (const float*)d_in[10]; const float* bmu = (const float*)d_in[11];

  char* ws = (char*)d_ws;
  const size_t WSZ = (size_t)DDIM * HDIM;  // 589824 elements per weight matrix
  unsigned short* wWq  = (unsigned short*)ws; ws += WSZ * 2;
  unsigned short* wWk  = (unsigned short*)ws; ws += WSZ * 2;
  unsigned short* wWv  = (unsigned short*)ws; ws += WSZ * 2;
  unsigned short* wWmu = (unsigned short*)ws; ws += WSZ * 2;
  unsigned short* wWvr = (unsigned short*)ws; ws += WSZ * 2;
  unsigned short* Qbf  = (unsigned short*)ws; ws += (size_t)BATCH * NQP * HDIM * 2;
  unsigned short* Kbf  = (unsigned short*)ws; ws += (size_t)BATCH * NKD * HDIM * 2;
  unsigned short* Vbf  = (unsigned short*)ws; ws += (size_t)BATCH * NKD * HDIM * 2;
  unsigned short* Abf  = (unsigned short*)ws; ws += (size_t)BATCH * NQP * HDIM * 2;

  const int nw = (int)WSZ;
  const dim3 cg((nw + 255) / 256), blk(256);
  cvt_f32_bf16_kernel<<<cg, blk, 0, stream>>>(Wq,   wWq,  nw);
  cvt_f32_bf16_kernel<<<cg, blk, 0, stream>>>(Wk,   wWk,  nw);
  cvt_f32_bf16_kernel<<<cg, blk, 0, stream>>>(Wv,   wWv,  nw);
  cvt_f32_bf16_kernel<<<cg, blk, 0, stream>>>(Wmu,  wWmu, nw);
  cvt_f32_bf16_kernel<<<cg, blk, 0, stream>>>(Wvar, wWvr, nw);

  // Q = spatial @ Wq + bq   (bf16 out, NQP-padded rows left untouched)
  gemm_bias_kernel<true, true, false>
      <<<dim3(HDIM / 64, (NQ + 63) / 64, BATCH), blk, 0, stream>>>(
          spatial, (long long)NQ * DDIM, NQ, wWq, bq, Qbf,
          (long long)NQP * HDIM, NQ, HDIM, DDIM);
  // K = text @ Wk + bk
  gemm_bias_kernel<true, true, false>
      <<<dim3(HDIM / 64, NKD / 64, BATCH), blk, 0, stream>>>(
          text, (long long)NKD * DDIM, NKD, wWk, bk, Kbf,
          (long long)NKD * HDIM, NKD, HDIM, DDIM);
  // V = text @ Wv + bv
  gemm_bias_kernel<true, true, false>
      <<<dim3(HDIM / 64, NKD / 64, BATCH), blk, 0, stream>>>(
          text, (long long)NKD * DDIM, NKD, wWv, bv, Vbf,
          (long long)NKD * HDIM, NKD, HDIM, DDIM);

  // fused attention
  const int attn_smem = 16 * 768 * 2 + 16 * 1032 * 4 + 16 * 1032 * 2 + 16 * 16 * 4;
  attn_kernel<<<dim3(NQP / 16, BATCH), blk, attn_smem, stream>>>(Qbf, Kbf, Vbf,
                                                                 Abf);

  // mu = attn @ Wmu + bmu (f32)
  gemm_bias_kernel<false, false, false>
      <<<dim3(DDIM / 64, (NQ + 63) / 64, BATCH), blk, 0, stream>>>(
          Abf, (long long)NQP * HDIM, NQP, wWmu, bmu, (float*)d_out,
          (long long)NQ * DDIM, NQ, DDIM, HDIM);
  // var = |attn @ Wvar + bvar| (f32, second output)
  float* outVar = (float*)d_out + (size_t)BATCH * NQ * DDIM;
  gemm_bias_kernel<false, false, true>
      <<<dim3(DDIM / 64, (NQ + 63) / 64, BATCH), blk, 0, stream>>>(
          Abf, (long long)NQP * HDIM, NQP, wWvr, bvar, outVar,
          (long long)NQ * DDIM, NQ, DDIM, HDIM);
}
